// PointNet2_45861660787348
// MI455X (gfx1250) — matrix operand order repaired and checked by reading, not compiled, not run on hardware
//
#include <hip/hip_runtime.h>
#include <hip/hip_bf16.h>
#include <cstdint>
#include <cstddef>

// ---------------------------------------------------------------------------
// PointNet++ (MSG) part-seg forward for MI455X / gfx1250.
// All MLP layers run through V_WMMA_F32_16X16X32_F16 (fp16 in, fp32 acc).
// Each wave computes 16 rows x (CT*16) cols, reusing one A fragment across
// CT column tiles; all CT B fragments are loaded before the WMMA chain so
// the split load counters can stagger instead of draining to 0 per WMMA.
// Point ops (FPS / ball-query / 3-NN / pool) are VALU/LDS kernels.
// ---------------------------------------------------------------------------

typedef __attribute__((ext_vector_type(16))) _Float16 v16h;
typedef __attribute__((ext_vector_type(8)))  _Float16 v8h;
typedef __attribute__((ext_vector_type(8)))  float    v8f;

#define BB 16
#define NN 4096

// ---------------------------------------------------------------------------
// Elementwise helpers
// ---------------------------------------------------------------------------
__global__ void bcn_to_bnc_kernel(const float* __restrict__ in,
                                  float* __restrict__ out,
                                  int B, int C, int N) {
    int t = blockIdx.x * blockDim.x + threadIdx.x;
    if (t >= B * C * N) return;
    int n = t % N;
    int c = (t / N) % C;
    int b = t / (N * C);
    out[((size_t)b * N + n) * C + c] = in[t];
}

__global__ void pack_w_kernel(const float* __restrict__ w,
                              _Float16* __restrict__ wh,
                              int O, int C, int Opack, int ldw) {
    int t = blockIdx.x * blockDim.x + threadIdx.x;
    if (t >= Opack * ldw) return;
    int o = t / ldw, c = t % ldw;
    wh[t] = (o < O && c < C) ? (_Float16)w[(size_t)o * C + c] : (_Float16)0.f;
}

// ---------------------------------------------------------------------------
// Farthest point sampling: one block per batch; dist kept in LDS.
// Records the carry-in `far` each step (matches reference scan semantics).
// ---------------------------------------------------------------------------
__global__ void __launch_bounds__(256)
fps_kernel(const float* __restrict__ xyz, int* __restrict__ idx, int N, int M) {
    __shared__ float sdist[4096];
    __shared__ float sv[256];
    __shared__ int   si[256];
    int b = blockIdx.x, tid = threadIdx.x;
    const float* xb = xyz + (size_t)b * N * 3;
    for (int n = tid; n < N; n += 256) sdist[n] = 1e10f;
    __syncthreads();
    int far = 0;
    for (int i = 0; i < M; ++i) {
        if (tid == 0) idx[b * M + i] = far;
        float cx = xb[far * 3 + 0], cy = xb[far * 3 + 1], cz = xb[far * 3 + 2];
        float lmax = -1.0f; int larg = 0;
        for (int n = tid; n < N; n += 256) {
            float dx = xb[n * 3 + 0] - cx;
            float dy = xb[n * 3 + 1] - cy;
            float dz = xb[n * 3 + 2] - cz;
            float d  = dx * dx + dy * dy + dz * dz;
            float nd = fminf(sdist[n], d);
            sdist[n] = nd;
            if (nd > lmax) { lmax = nd; larg = n; }
        }
        sv[tid] = lmax; si[tid] = larg;
        __syncthreads();
        for (int s = 128; s > 0; s >>= 1) {
            if (tid < s) {
                float v2 = sv[tid + s]; int i2 = si[tid + s];
                if (v2 > sv[tid] || (v2 == sv[tid] && i2 < si[tid])) {
                    sv[tid] = v2; si[tid] = i2;
                }
            }
            __syncthreads();
        }
        far = si[0];
        __syncthreads();
    }
}

__global__ void gather3_kernel(const float* __restrict__ xyz,
                               const int* __restrict__ idx,
                               float* __restrict__ out, int N, int M, int total) {
    int t = blockIdx.x * blockDim.x + threadIdx.x;
    if (t >= total) return;
    int c = t % 3;
    int j = (t / 3) % M;
    int b = t / (3 * M);
    out[t] = xyz[((size_t)b * N + idx[b * M + j]) * 3 + c];
}

// ---------------------------------------------------------------------------
// Ball query: one wave per centroid; ascending first-K in-radius indices,
// padded with the first hit (reference semantics).
// ---------------------------------------------------------------------------
__global__ void __launch_bounds__(128)
ball_query_kernel(const float* __restrict__ xyz, const float* __restrict__ cent,
                  int* __restrict__ out, int BM, int N, int M, int K, float r2) {
    int wv = blockIdx.x * 4 + (threadIdx.x >> 5);
    if (wv >= BM) return;
    int b = wv / M;
    int lane = threadIdx.x & 31;
    const float* xb = xyz + (size_t)b * N * 3;
    const float* c3 = cent + (size_t)wv * 3;
    float cx = c3[0], cy = c3[1], cz = c3[2];
    int* o = out + (size_t)wv * K;
    int cnt = 0, first = -1;
    for (int base = 0; base < N && cnt < K; base += 32) {
        int n = base + lane;
        bool inR = false;
        if (n < N) {
            float dx = xb[n * 3 + 0] - cx;
            float dy = xb[n * 3 + 1] - cy;
            float dz = xb[n * 3 + 2] - cz;
            inR = (dx * dx + dy * dy + dz * dz) <= r2;
        }
        unsigned mask = (unsigned)__ballot(inR);
        while (mask && cnt < K) {
            int bit = __builtin_ctz(mask);
            mask &= mask - 1;
            int p = base + bit;
            if (first < 0) first = p;
            if (lane == 0) o[cnt] = p;
            cnt++;
        }
    }
    if (first < 0) first = 0;
    for (int k = cnt + lane; k < K; k += 32) o[k] = first;
}

// ---------------------------------------------------------------------------
// Grouping: build fp16 rows [gx(3) | feats(Cf) | zero-pad] of stride Cpad.
// ---------------------------------------------------------------------------
__global__ void group_f32_kernel(const float* __restrict__ xb,
                                 const float* __restrict__ cb,
                                 const float* __restrict__ fb,
                                 const int* __restrict__ ball,
                                 _Float16* __restrict__ g,
                                 int M, int K, int Cf, int Cpad) {
    int rid = blockIdx.x * blockDim.x + threadIdx.x;
    if (rid >= M * K) return;
    int j = rid / K;
    int gi = ball[rid];
    _Float16* go = g + (size_t)rid * Cpad;
    for (int c = 0; c < 3; ++c)  go[c] = (_Float16)(xb[gi * 3 + c] - cb[j * 3 + c]);
    for (int c = 0; c < Cf; ++c) go[3 + c] = (_Float16)fb[(size_t)gi * Cf + c];
    for (int c = 3 + Cf; c < Cpad; ++c) go[c] = (_Float16)0.f;
}

__global__ void group_f16_kernel(const float* __restrict__ xb,
                                 const float* __restrict__ cb,
                                 const _Float16* __restrict__ fb,
                                 const int* __restrict__ ball,
                                 _Float16* __restrict__ g,
                                 int M, int K, int Cf, int Cpad) {
    int rid = blockIdx.x * blockDim.x + threadIdx.x;
    if (rid >= M * K) return;
    int j = rid / K;
    int gi = ball[rid];
    _Float16* go = g + (size_t)rid * Cpad;
    for (int c = 0; c < 3; ++c)  go[c] = (_Float16)(xb[gi * 3 + c] - cb[j * 3 + c]);
    for (int c = 0; c < Cf; ++c) go[3 + c] = fb[(size_t)gi * Cf + c];
    for (int c = 3 + Cf; c < Cpad; ++c) go[c] = (_Float16)0.f;
}

// ---------------------------------------------------------------------------
// WMMA GEMM: Y[p, o] = act(sum_c X[p,c] * W[o,c] + bias[o]).
// X: P x ldx fp16 (ldx multiple of 32), W: packed Opack x Cpad fp16.
// A fragment: lanes 0-15 row M=lane, K 0..15;
//             lanes 16-31 same rows, K 16..31.
// B fragment: lane = column; lanes 0-15 K 0..15, lanes 16-31 K 16..31.
// ---------------------------------------------------------------------------
template <int CT>
__global__ void __launch_bounds__(128)
gemm_wmma_kernel(const _Float16* __restrict__ X, const _Float16* __restrict__ W,
                 const float* __restrict__ bias, _Float16* __restrict__ Y,
                 float* __restrict__ Yf, int P, int Cpad, int O,
                 int ldx, int ldy, int relu, int n_per, int o_out) {
    int wave = blockIdx.x * 4 + (threadIdx.x >> 5);
    int grps = ldy / (16 * CT);           // column-tile groups
    int rowTile = wave / grps;
    int grp = wave % grps;
    if (rowTile * 16 >= P) return;
    int lane = threadIdx.x & 31;
    int lh = lane >> 4;                   // half-wave select
    int lm = lane & 15;
    int row = rowTile * 16 + lm;          // A row held by this lane

    const _Float16* xrow = X + (size_t)row * ldx;
    const _Float16* wbase = W + (size_t)(grp * CT * 16 + lm) * Cpad + lh * 16;
    v8f acc[CT];
#pragma unroll
    for (int ct = 0; ct < CT; ++ct) acc[ct] = (v8f){};

    // Prologue: A fragment of the first K-step.
    v8h alo = *(const v8h*)(xrow + lh * 16);
    v8h ahi = *(const v8h*)(xrow + 8 + lh * 16);
    for (int kk = 0; kk < Cpad; kk += 32) {
        // Load ALL B fragments for this K-step before the WMMA chain.
        v16h bf[CT];
#pragma unroll
        for (int ct = 0; ct < CT; ++ct)
            bf[ct] = *(const v16h*)(wbase + (size_t)ct * 16 * Cpad + kk);
        v16h a = __builtin_shufflevector(alo, ahi,
                                         0, 1, 2, 3, 4, 5, 6, 7,
                                         8, 9, 10, 11, 12, 13, 14, 15);
        // Hoist next iteration's A loads so they overlap the WMMA chain.
        if (kk + 32 < Cpad) {
            __builtin_prefetch(xrow + kk + 160, 0, 1);
            alo = *(const v8h*)(xrow + kk + 32 + lh * 16);
            ahi = *(const v8h*)(xrow + kk + 40 + lh * 16);
        }
#pragma unroll
        for (int ct = 0; ct < CT; ++ct)
            acc[ct] = __builtin_amdgcn_wmma_f32_16x16x32_f16(
                false, a, false, bf[ct], (short)0, acc[ct], false, false);
    }
#pragma unroll
    for (int ct = 0; ct < CT; ++ct) {
        int col = grp * CT * 16 + ct * 16 + lm;     // output column (this lane)
        float bv = (col < O) ? bias[col] : 0.0f;
#pragma unroll
        for (int r = 0; r < 8; ++r) {
            int orow = rowTile * 16 + r + lh * 8;   // D: VGPR r -> M=r / M=8+r
            float v = acc[ct][r] + bv;
            if (relu) v = v > 0.f ? v : 0.f;
            if (Y) {
                Y[(size_t)orow * ldy + col] = (col < O) ? (_Float16)v : (_Float16)0.f;
            }
            if (Yf && col < O) {
                int bb = orow / n_per, nn = orow % n_per;
                Yf[((size_t)bb * o_out + col) * n_per + nn] = v;  // (B,O,N)
            }
        }
    }
}

// ---------------------------------------------------------------------------
// Max-pool over K grouped samples (post-ReLU values, all >= 0).
// ---------------------------------------------------------------------------
__global__ void pool_kernel(const _Float16* __restrict__ in,
                            _Float16* __restrict__ out,
                            int M, int K, int ld, int C, int outLd, int colOff) {
    int t = blockIdx.x * blockDim.x + threadIdx.x;
    if (t >= M * C) return;
    int j = t / C, c = t % C;
    const _Float16* p = in + (size_t)j * K * ld + c;
    float m = -3.0e38f;
    for (int k = 0; k < K; ++k) {
        float v = (float)p[(size_t)k * ld];
        m = v > m ? v : m;
    }
    out[(size_t)j * outLd + colOff + c] = (_Float16)m;
}

// ---------------------------------------------------------------------------
// Concats / interpolation
// ---------------------------------------------------------------------------
__global__ void concat_gsa_kernel(const float* __restrict__ p2,
                                  const _Float16* __restrict__ f2h,
                                  _Float16* __restrict__ x, int total) {
    int t = blockIdx.x * blockDim.x + threadIdx.x;
    if (t >= total) return;
    int c = t % 544, row = t / 544;
    _Float16 v;
    if (c < 3)        v = (_Float16)p2[(size_t)row * 3 + c];
    else if (c < 515) v = f2h[(size_t)row * 512 + (c - 3)];
    else              v = (_Float16)0.f;
    x[t] = v;
}

__global__ void concat_fp1_kernel(const _Float16* __restrict__ f2h,
                                  const _Float16* __restrict__ f3h,
                                  _Float16* __restrict__ x, int total) {
    int t = blockIdx.x * blockDim.x + threadIdx.x;
    if (t >= total) return;
    int c = t % 1536, row = t / 1536, b = row / 128;
    x[t] = (c < 512) ? f2h[(size_t)row * 512 + c]
                     : f3h[(size_t)b * 1024 + (c - 512)];
}

__global__ void nn3_kernel(const float* __restrict__ dst,
                           const float* __restrict__ src,
                           int* __restrict__ id3, float* __restrict__ w3,
                           int BN, int Nd, int Ns) {
    int t = blockIdx.x * blockDim.x + threadIdx.x;
    if (t >= BN) return;
    int b = t / Nd;
    const float* d  = dst + (size_t)t * 3;
    const float* sb = src + (size_t)b * Ns * 3;
    float px = d[0], py = d[1], pz = d[2];
    float bd0 = 3e38f, bd1 = 3e38f, bd2 = 3e38f;
    int bi0 = 0, bi1 = 0, bi2 = 0;
    for (int s = 0; s < Ns; ++s) {
        float dx = sb[s * 3 + 0] - px;
        float dy = sb[s * 3 + 1] - py;
        float dz = sb[s * 3 + 2] - pz;
        float dd = dx * dx + dy * dy + dz * dz;
        if (dd < bd0)      { bd2 = bd1; bi2 = bi1; bd1 = bd0; bi1 = bi0; bd0 = dd; bi0 = s; }
        else if (dd < bd1) { bd2 = bd1; bi2 = bi1; bd1 = dd; bi1 = s; }
        else if (dd < bd2) { bd2 = dd; bi2 = s; }
    }
    float w0 = 1.f / (bd0 + 1e-8f);
    float w1 = 1.f / (bd1 + 1e-8f);
    float w2 = 1.f / (bd2 + 1e-8f);
    float ws = w0 + w1 + w2;
    id3[t * 3 + 0] = bi0; id3[t * 3 + 1] = bi1; id3[t * 3 + 2] = bi2;
    w3[t * 3 + 0] = w0 / ws; w3[t * 3 + 1] = w1 / ws; w3[t * 3 + 2] = w2 / ws;
}

__global__ void concat_fp2_kernel(const _Float16* __restrict__ f1h,
                                  const _Float16* __restrict__ f2p,
                                  const int* __restrict__ id3,
                                  const float* __restrict__ w3,
                                  _Float16* __restrict__ x, int total) {
    int t = blockIdx.x * blockDim.x + threadIdx.x;
    if (t >= total) return;
    int c = t % 576, row = t / 576, b = row / 512;
    if (c < 320) {
        x[t] = f1h[(size_t)row * 320 + c];
    } else {
        int cc = c - 320;
        float a = 0.f;
        for (int q = 0; q < 3; ++q)
            a += w3[row * 3 + q] *
                 (float)f2p[((size_t)b * 128 + id3[row * 3 + q]) * 256 + cc];
        x[t] = (_Float16)a;
    }
}

__global__ void concat_fp3_kernel(const int* __restrict__ cls_id,
                                  const float* __restrict__ features,
                                  const float* __restrict__ points,
                                  const _Float16* __restrict__ f1p,
                                  const int* __restrict__ id3,
                                  const float* __restrict__ w3,
                                  _Float16* __restrict__ x, int total) {
    int t = blockIdx.x * blockDim.x + threadIdx.x;
    if (t >= total) return;
    int c = t % 160, row = t / 160;
    int b = row / 4096, n = row % 4096;
    _Float16 v;
    if (c < 16) {
        v = (_Float16)((c == cls_id[b]) ? 1.f : 0.f);
    } else if (c < 19) {
        v = (_Float16)features[((size_t)b * 3 + (c - 16)) * 4096 + n];
    } else if (c < 22) {
        v = (_Float16)points[((size_t)b * 3 + (c - 19)) * 4096 + n];
    } else if (c < 150) {
        int cc = c - 22;
        float a = 0.f;
        for (int q = 0; q < 3; ++q)
            a += w3[row * 3 + q] *
                 (float)f1p[((size_t)b * 512 + id3[row * 3 + q]) * 128 + cc];
        v = (_Float16)a;
    } else {
        v = (_Float16)0.f;
    }
    x[t] = v;
}

// ---------------------------------------------------------------------------
// Host orchestration
// ---------------------------------------------------------------------------
struct LayerDef { int wi; int O; int C; };

extern "C" void kernel_launch(void* const* d_in, const int* in_sizes, int n_in,
                              void* d_out, int out_size, void* d_ws, size_t ws_size,
                              hipStream_t stream) {
    (void)in_sizes; (void)n_in; (void)out_size; (void)ws_size;
    const float* points   = (const float*)d_in[0];   // (B,3,N)
    const float* features = (const float*)d_in[1];   // (B,3,N)
    const int*   class_id = (const int*)d_in[2];
    auto Wf = [&](int i) { return (const float*)d_in[i]; };
    auto al32 = [](int x) { return (x + 31) & ~31; };

    // Params flattened by the harness in JAX pytree order (dict keys sorted):
    // cls, fp1, fp2, fp3, gsa, sa1(3 branches), sa2(2 branches); (w,b) pairs.
    static const LayerDef L[26] = {
        {3, 128, 128}, {5, 50, 128},                       // cls     [0..1]
        {7, 256, 1536}, {9, 256, 256},                     // fp1     [2..3]
        {11, 256, 576}, {13, 128, 256},                    // fp2     [4..5]
        {15, 128, 150}, {17, 128, 128},                    // fp3     [6..7]
        {19, 256, 515}, {21, 512, 256}, {23, 1024, 512},   // gsa     [8..10]
        {25, 32, 6},  {27, 32, 32},  {29, 64, 32},         // sa1 b0  [11..13]
        {31, 64, 6},  {33, 64, 64},  {35, 128, 64},        // sa1 b1  [14..16]
        {37, 64, 6},  {39, 96, 64},  {41, 128, 96},        // sa1 b2  [17..19]
        {43, 128, 323}, {45, 128, 128}, {47, 256, 128},    // sa2 b0  [20..22]
        {49, 128, 323}, {51, 196, 128}, {53, 256, 196},    // sa2 b1  [23..25]
    };

    // --- workspace bump allocator -------------------------------------------
    size_t off = 0;
    auto alloc = [&](size_t bytes) -> void* {
        void* p = (char*)d_ws + off;
        off = (off + bytes + 255) & ~(size_t)255;
        return p;
    };
    float*    xyz   = (float*)alloc((size_t)BB * NN * 3 * 4);
    float*    fnlc  = (float*)alloc((size_t)BB * NN * 3 * 4);
    int*      idx1  = (int*)alloc((size_t)BB * 512 * 4);
    float*    nxyz1 = (float*)alloc((size_t)BB * 512 * 3 * 4);
    int*      idx2  = (int*)alloc((size_t)BB * 128 * 4);
    float*    nxyz2 = (float*)alloc((size_t)BB * 128 * 3 * 4);
    int*      ball  = (int*)alloc((size_t)BB * 512 * 128 * 4);
    _Float16* f1h   = (_Float16*)alloc((size_t)BB * 512 * 320 * 2);
    _Float16* f2h   = (_Float16*)alloc((size_t)BB * 128 * 512 * 2);
    _Float16* f3h   = (_Float16*)alloc((size_t)BB * 1024 * 2);
    _Float16* f2p   = (_Float16*)alloc((size_t)BB * 128 * 256 * 2);
    _Float16* f1p   = (_Float16*)alloc((size_t)BB * 512 * 128 * 2);
    int*      id3   = (int*)alloc((size_t)BB * 4096 * 3 * 4);
    float*    w3    = (float*)alloc((size_t)BB * 4096 * 3 * 4);
    _Float16* gA    = (_Float16*)alloc((size_t)65536 * 192 * 2);
    _Float16* gB    = (_Float16*)alloc((size_t)65536 * 192 * 2);

    // --- pack all weights to fp16 (O padded to 32, C padded to 32) ----------
    _Float16* wp[26];
    for (int i = 0; i < 26; ++i) {
        int O = L[i].O, C = L[i].C;
        int Op = al32(O), ld = al32(C);
        wp[i] = (_Float16*)alloc((size_t)Op * ld * 2);
        int tot = Op * ld;
        pack_w_kernel<<<(tot + 255) / 256, 256, 0, stream>>>(Wf(L[i].wi), wp[i], O, C, Op, ld);
    }

    auto gemm = [&](const _Float16* X, int li, _Float16* Y, float* Yf,
                    int P, int ldx, int relu, int ldy, int n_per, int o_out) {
        int Cpad = al32(L[li].C);
        if (ldy % 64 == 0) {
            int tiles = (P / 16) * (ldy / 64);
            gemm_wmma_kernel<4><<<(tiles + 3) / 4, 128, 0, stream>>>(
                X, wp[li], Wf(L[li].wi + 1), Y, Yf, P, Cpad, L[li].O, ldx, ldy,
                relu, n_per, o_out);
        } else {  // all ldy are multiples of 32
            int tiles = (P / 16) * (ldy / 32);
            gemm_wmma_kernel<2><<<(tiles + 3) / 4, 128, 0, stream>>>(
                X, wp[li], Wf(L[li].wi + 1), Y, Yf, P, Cpad, L[li].O, ldx, ldy,
                relu, n_per, o_out);
        }
    };

    // --- layout transforms ---------------------------------------------------
    bcn_to_bnc_kernel<<<(BB * 3 * NN + 255) / 256, 256, 0, stream>>>(points, xyz, BB, 3, NN);
    bcn_to_bnc_kernel<<<(BB * 3 * NN + 255) / 256, 256, 0, stream>>>(features, fnlc, BB, 3, NN);

    // === SA1 (MSG): 4096 -> 512 centroids ===================================
    fps_kernel<<<BB, 256, 0, stream>>>(xyz, idx1, NN, 512);
    gather3_kernel<<<(BB * 512 * 3 + 255) / 256, 256, 0, stream>>>(xyz, idx1, nxyz1, NN, 512, BB * 512 * 3);
    {
        const int   Ks[3]   = {32, 64, 128};
        const float rs[3]   = {0.1f, 0.2f, 0.4f};
        const int   l0s[3]  = {11, 14, 17};
        const int   coff[3] = {0, 64, 192};
        for (int br = 0; br < 3; ++br) {
            int K = Ks[br];
            ball_query_kernel<<<(BB * 512 + 3) / 4, 128, 0, stream>>>(
                xyz, nxyz1, ball, BB * 512, NN, 512, K, rs[br] * rs[br]);
            int l0 = l0s[br];
            int ld0 = al32(L[l0].O), ld1 = al32(L[l0 + 1].O), ld2 = al32(L[l0 + 2].O);
            for (int b = 0; b < BB; ++b) {
                int rows = 512 * K;
                group_f32_kernel<<<(rows + 255) / 256, 256, 0, stream>>>(
                    xyz + (size_t)b * NN * 3, nxyz1 + (size_t)b * 512 * 3,
                    fnlc + (size_t)b * NN * 3, ball + (size_t)b * 512 * K,
                    gA, 512, K, 3, 32);
                gemm(gA, l0,     gB, nullptr, rows, 32,  1, ld0, 0, 0);
                gemm(gB, l0 + 1, gA, nullptr, rows, ld0, 1, ld1, 0, 0);
                gemm(gA, l0 + 2, gB, nullptr, rows, ld1, 1, ld2, 0, 0);
                pool_kernel<<<(512 * L[l0 + 2].O + 255) / 256, 256, 0, stream>>>(
                    gB, f1h + (size_t)b * 512 * 320, 512, K, ld2, L[l0 + 2].O, 320, coff[br]);
            }
        }
    }

    // === SA2 (MSG): 512 -> 128 centroids ====================================
    fps_kernel<<<BB, 256, 0, stream>>>(nxyz1, idx2, 512, 128);
    gather3_kernel<<<(BB * 128 * 3 + 255) / 256, 256, 0, stream>>>(nxyz1, idx2, nxyz2, 512, 128, BB * 128 * 3);
    {
        const int   Ks[2]   = {64, 128};
        const float rs[2]   = {0.4f, 0.8f};
        const int   l0s[2]  = {20, 23};
        const int   coff[2] = {0, 256};
        for (int br = 0; br < 2; ++br) {
            int K = Ks[br];
            ball_query_kernel<<<(BB * 128 + 3) / 4, 128, 0, stream>>>(
                nxyz1, nxyz2, ball, BB * 128, 512, 128, K, rs[br] * rs[br]);
            int l0 = l0s[br];
            int ld0 = al32(L[l0].O), ld1 = al32(L[l0 + 1].O), ld2 = al32(L[l0 + 2].O);
            for (int b = 0; b < BB; ++b) {
                int rows = 128 * K;
                group_f16_kernel<<<(rows + 255) / 256, 256, 0, stream>>>(
                    nxyz1 + (size_t)b * 512 * 3, nxyz2 + (size_t)b * 128 * 3,
                    f1h + (size_t)b * 512 * 320, ball + (size_t)b * 128 * K,
                    gA, 128, K, 320, 352);
                gemm(gA, l0,     gB, nullptr, rows, 352, 1, ld0, 0, 0);
                gemm(gB, l0 + 1, gA, nullptr, rows, ld0, 1, ld1, 0, 0);
                gemm(gA, l0 + 2, gB, nullptr, rows, ld1, 1, ld2, 0, 0);
                pool_kernel<<<(128 * L[l0 + 2].O + 255) / 256, 256, 0, stream>>>(
                    gB, f2h + (size_t)b * 128 * 512, 128, K, ld2, L[l0 + 2].O, 512, coff[br]);
            }
        }
    }

    // === GSA (group-all): 128 pts -> 1 global feature =======================
    concat_gsa_kernel<<<(BB * 128 * 544 + 255) / 256, 256, 0, stream>>>(nxyz2, f2h, gA, BB * 128 * 544);
    gemm(gA, 8,  gB, nullptr, BB * 128, 544, 1, 256, 0, 0);
    gemm(gB, 9,  gA, nullptr, BB * 128, 256, 1, 512, 0, 0);
    gemm(gA, 10, gB, nullptr, BB * 128, 512, 1, 1024, 0, 0);
    for (int b = 0; b < BB; ++b)
        pool_kernel<<<(1024 + 255) / 256, 256, 0, stream>>>(
            gB + (size_t)b * 128 * 1024, f3h + (size_t)b * 1024, 1, 128, 1024, 1024, 1024, 0);

    // === FP1: broadcast global feature to 128 pts ===========================
    concat_fp1_kernel<<<(BB * 128 * 1536 + 255) / 256, 256, 0, stream>>>(f2h, f3h, gA, BB * 128 * 1536);
    gemm(gA, 2, gB,  nullptr, BB * 128, 1536, 1, 256, 0, 0);
    gemm(gB, 3, f2p, nullptr, BB * 128, 256,  1, 256, 0, 0);

    // === FP2: 3-NN interpolate 128 -> 512 ===================================
    nn3_kernel<<<(BB * 512 + 255) / 256, 256, 0, stream>>>(nxyz1, nxyz2, id3, w3, BB * 512, 512, 128);
    concat_fp2_kernel<<<(BB * 512 * 576 + 255) / 256, 256, 0, stream>>>(f1h, f2p, id3, w3, gA, BB * 512 * 576);
    gemm(gA, 4, gB,  nullptr, BB * 512, 576, 1, 256, 0, 0);
    gemm(gB, 5, f1p, nullptr, BB * 512, 256, 1, 128, 0, 0);

    // === FP3: 3-NN interpolate 512 -> 4096, concat one-hot + raw inputs =====
    nn3_kernel<<<(BB * 4096 + 255) / 256, 256, 0, stream>>>(xyz, nxyz1, id3, w3, BB * 4096, 4096, 512);
    concat_fp3_kernel<<<(BB * 4096 * 160 + 255) / 256, 256, 0, stream>>>(
        class_id, features, points, f1p, id3, w3, gA, BB * 4096 * 160);
    gemm(gA, 6, gB, nullptr, BB * 4096, 160, 1, 128, 0, 0);
    gemm(gB, 7, gA, nullptr, BB * 4096, 128, 1, 128, 0, 0);

    // === classifier head: 128 -> 128 (relu) -> 50, output (B,50,N) fp32 =====
    gemm(gA, 0, gB, nullptr, BB * 4096, 128, 1, 128, 0, 0);
    gemm(gB, 1, nullptr, (float*)d_out, BB * 4096, 128, 0, 64, 4096, 50);
}
